// MMSingleStreamBlockAudioI2V_2396591751869
// MI455X (gfx1250) — compile-verified, matrix operand order checked by reading, not run on hardware
//
#include <hip/hip_runtime.h>
#include <hip/hip_bf16.h>

// ---------------- constants (match reference shapes) ----------------
#define HID   3072
#define HEADS 24
#define HD    128
#define MLP   12288
#define L     2048
#define L_IMG 1920
#define FFN   384
#define W1N   (3*HID+MLP)   // 21504
#define K2    (HID+MLP)     // 15360
#define L_AUD 128
#define AQN   (3*HID)       // 9216

typedef __attribute__((ext_vector_type(16))) _Float16 v16h;
typedef __attribute__((ext_vector_type(8)))  float    v8f;

#define WMMA_F16(a, b, c) \
  __builtin_amdgcn_wmma_f32_16x16x32_f16(false, (a), false, (b), (short)0, (c), false, false)

// K-dim mapping of element-pair p (0..7) for 16-bit WMMA fragments
// (ISA 7.12.2: VGPR j lanes0-15 K=2j,2j+1 ; lanes16-31 K=2j+8,... ; VGPR4-7 +16)
__device__ __forceinline__ int kpair_base(int p, int half) {
  return 2 * (p & 3) + 8 * half + 16 * (p >> 2);
}

__device__ __forceinline__ float blockReduceSum(float v) {
  __shared__ float sh[33];
  int lane = threadIdx.x & 31;
  int wid  = threadIdx.x >> 5;
  #pragma unroll
  for (int o = 16; o > 0; o >>= 1) v += __shfl_xor(v, o, 32);
  __syncthreads();
  if (lane == 0) sh[wid] = v;
  __syncthreads();
  float r = 0.f;
  int nw = (blockDim.x + 31) >> 5;
  for (int i = 0; i < nw; i++) r += sh[i];
  return r;
}

// ---------------- small elementwise / reduction kernels ----------------

// mod = silu(vec) @ mod_w + mod_b ; trm = silu(trv) @ mod_w + mod_b
__global__ void mod_kernel(const float* __restrict__ vec, const float* __restrict__ trv,
                           const float* __restrict__ mw, const float* __restrict__ mb,
                           float* __restrict__ mod, float* __restrict__ trm) {
  int col = blockIdx.x * blockDim.x + threadIdx.x;   // 0..9215
  float s0 = 0.f, s1 = 0.f;
  for (int k = 0; k < HID; ++k) {
    float w = mw[(size_t)k * AQN + col];
    float a = vec[k]; a = a / (1.f + __expf(-a));
    float b = trv[k]; b = b / (1.f + __expf(-b));
    s0 += a * w; s1 += b * w;
  }
  mod[col] = s0 + mb[col];
  trm[col] = s1 + mb[col];
}

// xm = LN(x) * (1+scale) + shift   (token-replace mod for rows < FFN)
__global__ void ln_mod_kernel(const float* __restrict__ x, const float* __restrict__ mod,
                              const float* __restrict__ trm, float* __restrict__ xm) {
  int row = blockIdx.x;
  const float* xr = x + (size_t)row * HID;
  float s = 0.f;
  for (int i = threadIdx.x; i < HID; i += blockDim.x) s += xr[i];
  float mean = blockReduceSum(s) * (1.f / HID);
  float v = 0.f;
  for (int i = threadIdx.x; i < HID; i += blockDim.x) { float d = xr[i] - mean; v += d * d; }
  float var = blockReduceSum(v) * (1.f / HID);
  float rstd = rsqrtf(var + 1e-6f);
  const float* mv = (row < FFN) ? trm : mod;   // [shift | scale | gate]
  float* o = xm + (size_t)row * HID;
  for (int i = threadIdx.x; i < HID; i += blockDim.x) {
    float xn = (xr[i] - mean) * rstd;
    o[i] = xn * (1.f + mv[HID + i]) + mv[i];
  }
}

// plain LayerNorm (audio path)
__global__ void ln_kernel(const float* __restrict__ src, float* __restrict__ dst, int cols) {
  int row = blockIdx.x;
  const float* xr = src + (size_t)row * cols;
  float s = 0.f;
  for (int i = threadIdx.x; i < cols; i += blockDim.x) s += xr[i];
  float mean = blockReduceSum(s) / (float)cols;
  float v = 0.f;
  for (int i = threadIdx.x; i < cols; i += blockDim.x) { float d = xr[i] - mean; v += d * d; }
  float var = blockReduceSum(v) / (float)cols;
  float rstd = rsqrtf(var + 1e-6f);
  float* o = dst + (size_t)row * cols;
  for (int i = threadIdx.x; i < cols; i += blockDim.x) o[i] = (xr[i] - mean) * rstd;
}

// per-(token,head) RMS 1/std over HD elements at src[tok*stride + col_off + head*HD]
__global__ void rms_stats_kernel(const float* __restrict__ src, size_t row_stride, int col_off,
                                 int nrows, float* __restrict__ out) {
  int idx = blockIdx.x * blockDim.x + threadIdx.x;
  if (idx >= nrows * HEADS) return;
  int tok = idx / HEADS, hd = idx % HEADS;
  const float* p = src + (size_t)tok * row_stride + col_off + hd * HD;
  float s = 0.f;
  for (int d = 0; d < HD; ++d) s += p[d] * p[d];
  out[idx] = rsqrtf(s * (1.f / HD) + 1e-6f);
}

// ---------------- fragment packing (f32 -> f16 WMMA fragment layout) ----------------

// A-matrix pack: src is M x K row-major f32. dst tile index = mt*(K/32)+kt, 512 halfs/tile.
__global__ void __launch_bounds__(32) pack_a_kernel(const float* __restrict__ src,
                                                    _Float16* __restrict__ dst, int K) {
  int Kt = K / 32;
  int tile = blockIdx.x;
  int mt = tile / Kt, kt = tile % Kt;
  int lane = threadIdx.x, half = lane >> 4;
  int m = mt * 16 + (lane & 15);
  const float* row = src + (size_t)m * K + kt * 32;
  v16h t;
  #pragma unroll
  for (int p = 0; p < 8; ++p) {
    int k = kpair_base(p, half);
    t[2 * p]     = (_Float16)row[k];
    t[2 * p + 1] = (_Float16)row[k + 1];
  }
  *(v16h*)(dst + (size_t)tile * 512 + lane * 16) = t;
}

// B-matrix pack: src is K x N row-major f32. dst tile index = nt*(K/32)+kt.
__global__ void __launch_bounds__(32) pack_b_kernel(const float* __restrict__ src,
                                                    _Float16* __restrict__ dst, int K, int N) {
  int Kt = K / 32;
  int tile = blockIdx.x;
  int nt = tile / Kt, kt = tile % Kt;
  int lane = threadIdx.x, half = lane >> 4;
  int n = nt * 16 + (lane & 15);
  v16h t;
  #pragma unroll
  for (int p = 0; p < 8; ++p) {
    int k = kt * 32 + kpair_base(p, half);
    t[2 * p]     = (_Float16)src[(size_t)k * N + n];
    t[2 * p + 1] = (_Float16)src[(size_t)(k + 1) * N + n];
  }
  *(v16h*)(dst + (size_t)tile * 512 + lane * 16) = t;
}

// RMS-norm (+ optional RoPE) + pack for Q (A-frag) or K (B-frag); identical index math.
// tile = ((head*rows16 + rt)*4 + c) ; c selects 32-wide chunk of HD.
__global__ void __launch_bounds__(32) pack_qk_kernel(const float* __restrict__ h, size_t row_stride,
                                                     int col_off, const float* __restrict__ rstd,
                                                     const float* __restrict__ nw,
                                                     const float* __restrict__ cf,
                                                     const float* __restrict__ sf,
                                                     int rows16, int rope_limit,
                                                     _Float16* __restrict__ dst) {
  int tile = blockIdx.x;
  int c = tile & 3; int rest = tile >> 2;
  int rt = rest % rows16; int head = rest / rows16;
  int lane = threadIdx.x, half = lane >> 4;
  int tok = rt * 16 + (lane & 15);
  float rs = rstd[tok * HEADS + head];
  const float* row = h + (size_t)tok * row_stride + col_off + head * HD;
  bool do_rope = tok < rope_limit;
  v16h t;
  #pragma unroll
  for (int p = 0; p < 8; ++p) {
    int d0 = c * 32 + kpair_base(p, half);
    #pragma unroll
    for (int e = 0; e < 2; ++e) {
      int d = d0 + e;
      float v = row[d] * rs * nw[d];
      if (do_rope) {
        int pd = d ^ 1;
        float pv = row[pd] * rs * nw[pd];
        float rot = (d & 1) ? pv : -pv;         // rot_half: (x0,x1)->(-x1,x0)
        v = v * cf[tok * HD + d] + rot * sf[tok * HD + d];
      }
      t[2 * p + e] = (_Float16)v;
    }
  }
  *(v16h*)(dst + (size_t)tile * 512 + lane * 16) = t;
}

// V pack as B-fragments of the P@V GEMM: tile = ((head*rows32 + kt)*8 + dt)
__global__ void __launch_bounds__(32) pack_v_kernel(const float* __restrict__ h, size_t row_stride,
                                                    int col_off, int rows32,
                                                    _Float16* __restrict__ dst) {
  int tile = blockIdx.x;
  int dt = tile & 7; int rest = tile >> 3;
  int kt = rest % rows32; int head = rest / rows32;
  int lane = threadIdx.x, half = lane >> 4;
  int dcol = dt * 16 + (lane & 15);
  v16h t;
  #pragma unroll
  for (int p = 0; p < 8; ++p) {
    int k0 = kt * 32 + kpair_base(p, half);
    #pragma unroll
    for (int e = 0; e < 2; ++e) {
      int key = k0 + e;
      t[2 * p + e] = (_Float16)h[(size_t)key * row_stride + col_off + head * HD + dcol];
    }
  }
  *(v16h*)(dst + (size_t)tile * 512 + lane * 16) = t;
}

// pack A for GEMM2: [ attn (3072) | gelu_tanh(mlp) (12288) ] per row
__global__ void __launch_bounds__(32) pack_a2_kernel(const float* __restrict__ attn,
                                                     const float* __restrict__ h,
                                                     _Float16* __restrict__ dst) {
  const int Kt = K2 / 32;   // 480
  int tile = blockIdx.x;
  int mt = tile / Kt, kt = tile % Kt;
  int lane = threadIdx.x, half = lane >> 4;
  int m = mt * 16 + (lane & 15);
  v16h t;
  #pragma unroll
  for (int p = 0; p < 8; ++p) {
    int k0 = kt * 32 + kpair_base(p, half);
    #pragma unroll
    for (int e = 0; e < 2; ++e) {
      int kk = k0 + e;
      float v;
      if (kk < HID) {
        v = attn[(size_t)m * HID + kk];
      } else {
        float xv = h[(size_t)m * W1N + 3 * HID + (kk - HID)];
        float x3 = xv * xv * xv;
        v = 0.5f * xv * (1.f + tanhf(0.7978845608f * (xv + 0.044715f * x3)));
      }
      t[2 * p + e] = (_Float16)v;
    }
  }
  *(v16h*)(dst + (size_t)tile * 512 + lane * 16) = t;
}

// ---------------- WMMA GEMM: C[M,N] = Apack x Bpack + bias ----------------
// grid = (N/64, M/32), block = 32. One wave computes a 32x64 tile (8 accumulators);
// fragments for step kt+1 are loaded into fresh registers before issuing step kt's
// 8 WMMAs, so loads overlap matrix math (software pipelining / double buffering).
__global__ void __launch_bounds__(32) wmma_gemm_kernel(const _Float16* __restrict__ Ap,
                                                       const _Float16* __restrict__ Bp,
                                                       const float* __restrict__ bias,
                                                       float* __restrict__ C, int N, int K) {
  const int Kt = K / 32;
  int nt0 = blockIdx.x * 4;
  int mt0 = blockIdx.y * 2;
  int lane = threadIdx.x, half = lane >> 4, nc = lane & 15;
  v8f zero = {0,0,0,0,0,0,0,0};
  v8f acc[8];
  #pragma unroll
  for (int i = 0; i < 8; ++i) acc[i] = zero;
  const _Float16* a0 = Ap + ((size_t)(mt0 + 0) * Kt) * 512 + lane * 16;
  const _Float16* a1 = Ap + ((size_t)(mt0 + 1) * Kt) * 512 + lane * 16;
  const _Float16* bb = Bp + (size_t)lane * 16;
  v16h ca0 = *(const v16h*)a0;
  v16h ca1 = *(const v16h*)a1;
  v16h cb0 = *(const v16h*)(bb + ((size_t)(nt0 + 0) * Kt) * 512);
  v16h cb1 = *(const v16h*)(bb + ((size_t)(nt0 + 1) * Kt) * 512);
  v16h cb2 = *(const v16h*)(bb + ((size_t)(nt0 + 2) * Kt) * 512);
  v16h cb3 = *(const v16h*)(bb + ((size_t)(nt0 + 3) * Kt) * 512);
  for (int kt = 1; kt < Kt; ++kt) {
    v16h na0 = *(const v16h*)(a0 + (size_t)kt * 512);
    v16h na1 = *(const v16h*)(a1 + (size_t)kt * 512);
    v16h nb0 = *(const v16h*)(bb + ((size_t)(nt0 + 0) * Kt + kt) * 512);
    v16h nb1 = *(const v16h*)(bb + ((size_t)(nt0 + 1) * Kt + kt) * 512);
    v16h nb2 = *(const v16h*)(bb + ((size_t)(nt0 + 2) * Kt + kt) * 512);
    v16h nb3 = *(const v16h*)(bb + ((size_t)(nt0 + 3) * Kt + kt) * 512);
    acc[0] = WMMA_F16(ca0, cb0, acc[0]);
    acc[1] = WMMA_F16(ca0, cb1, acc[1]);
    acc[2] = WMMA_F16(ca0, cb2, acc[2]);
    acc[3] = WMMA_F16(ca0, cb3, acc[3]);
    acc[4] = WMMA_F16(ca1, cb0, acc[4]);
    acc[5] = WMMA_F16(ca1, cb1, acc[5]);
    acc[6] = WMMA_F16(ca1, cb2, acc[6]);
    acc[7] = WMMA_F16(ca1, cb3, acc[7]);
    ca0 = na0; ca1 = na1; cb0 = nb0; cb1 = nb1; cb2 = nb2; cb3 = nb3;
  }
  acc[0] = WMMA_F16(ca0, cb0, acc[0]);
  acc[1] = WMMA_F16(ca0, cb1, acc[1]);
  acc[2] = WMMA_F16(ca0, cb2, acc[2]);
  acc[3] = WMMA_F16(ca0, cb3, acc[3]);
  acc[4] = WMMA_F16(ca1, cb0, acc[4]);
  acc[5] = WMMA_F16(ca1, cb1, acc[5]);
  acc[6] = WMMA_F16(ca1, cb2, acc[6]);
  acc[7] = WMMA_F16(ca1, cb3, acc[7]);
  #pragma unroll
  for (int mi = 0; mi < 2; ++mi) {
    #pragma unroll
    for (int j = 0; j < 4; ++j) {
      int col = (nt0 + j) * 16 + nc;
      float bv = bias[col];
      #pragma unroll
      for (int r = 0; r < 8; ++r) {
        int row = (mt0 + mi) * 16 + r + 8 * half;
        C[(size_t)row * N + col] = acc[mi * 4 + j][r] + bv;
      }
    }
  }
}

// ---------------- flash attention (self, block-diagonal mask via tile range) ----------------
// grid = (L/16, HEADS), block = 32. segments: [0,2016) and [2016,2048).
__global__ void __launch_bounds__(32) attn_kernel(const _Float16* __restrict__ qp,
                                                  const _Float16* __restrict__ kp,
                                                  const _Float16* __restrict__ vp,
                                                  float* __restrict__ attn) {
  __shared__ _Float16 plds[16 * 32];
  const int qt = blockIdx.x, head = blockIdx.y;
  const int lane = threadIdx.x, half = lane >> 4, nc = lane & 15;
  const float scale = 0.08838834764831845f;  // 128^-0.5
  v16h qf[4];
  const _Float16* qb = qp + (((size_t)head * (L / 16) + qt) * 4) * 512 + lane * 16;
  #pragma unroll
  for (int c = 0; c < 4; ++c) qf[c] = *(const v16h*)(qb + (size_t)c * 512);
  v8f zero = {0,0,0,0,0,0,0,0};
  v8f o[8];
  #pragma unroll
  for (int t = 0; t < 8; ++t) o[t] = zero;
  float mrow[8], lrow[8];
  #pragma unroll
  for (int r = 0; r < 8; ++r) { mrow[r] = -1e30f; lrow[r] = 0.f; }
  int ktb = (qt < 126) ? 0 : 126;
  int kte = (qt < 126) ? 126 : 128;
  for (int kt = ktb; kt < kte; kt += 2) {
    v8f s0 = zero, s1 = zero;
    #pragma unroll
    for (int c = 0; c < 4; ++c) {
      v16h b0 = *(const v16h*)(kp + (((size_t)head * (L / 16) + kt) * 4 + c) * 512 + lane * 16);
      v16h b1 = *(const v16h*)(kp + (((size_t)head * (L / 16) + kt + 1) * 4 + c) * 512 + lane * 16);
      s0 = WMMA_F16(qf[c], b0, s0);
      s1 = WMMA_F16(qf[c], b1, s1);
    }
    // issue V-fragment loads early: softmax VALU work below hides their latency
    const _Float16* vb = vp + (((size_t)head * (L / 32) + (kt >> 1)) * 8) * 512 + lane * 16;
    v16h vf[8];
    #pragma unroll
    for (int t = 0; t < 8; ++t) vf[t] = *(const v16h*)(vb + (size_t)t * 512);
    float corr[8];
    #pragma unroll
    for (int r = 0; r < 8; ++r) {
      float a0 = s0[r] * scale, a1 = s1[r] * scale;
      float mx = fmaxf(a0, a1);
      #pragma unroll
      for (int off = 1; off < 16; off <<= 1) mx = fmaxf(mx, __shfl_xor(mx, off, 32));
      float nm = fmaxf(mrow[r], mx);
      corr[r] = __expf(mrow[r] - nm);
      float p0 = __expf(a0 - nm), p1 = __expf(a1 - nm);
      float ps = p0 + p1;
      #pragma unroll
      for (int off = 1; off < 16; off <<= 1) ps += __shfl_xor(ps, off, 32);
      lrow[r] = lrow[r] * corr[r] + ps;
      mrow[r] = nm;
      plds[(r + 8 * half) * 32 + nc]      = (_Float16)p0;
      plds[(r + 8 * half) * 32 + nc + 16] = (_Float16)p1;
    }
    __syncthreads();
    #pragma unroll
    for (int t = 0; t < 8; ++t)
      #pragma unroll
      for (int r = 0; r < 8; ++r) o[t][r] *= corr[r];
    v16h pf;
    #pragma unroll
    for (int p = 0; p < 8; ++p) {
      int k = kpair_base(p, half);
      pf[2 * p]     = plds[nc * 32 + k];
      pf[2 * p + 1] = plds[nc * 32 + k + 1];
    }
    #pragma unroll
    for (int t = 0; t < 8; ++t) o[t] = WMMA_F16(pf, vf[t], o[t]);
    __syncthreads();
  }
  #pragma unroll
  for (int t = 0; t < 8; ++t) {
    int col = head * HD + t * 16 + nc;
    #pragma unroll
    for (int r = 0; r < 8; ++r) {
      int row = qt * 16 + r + 8 * half;
      attn[(size_t)row * HID + col] = o[t][r] / lrow[r];
    }
  }
}

// ---------------- audio cross-attention: iq x ak -> +face_mask*scale into attn ----------------
// grid = (L_IMG/16, HEADS), block = 32; 128 keys, no mask.
__global__ void __launch_bounds__(32) audio_attn_kernel(const _Float16* __restrict__ qp,
                                                        const _Float16* __restrict__ akp,
                                                        const _Float16* __restrict__ avp,
                                                        const float* __restrict__ fmask,
                                                        const float* __restrict__ ascale,
                                                        float* __restrict__ attn) {
  __shared__ _Float16 plds[16 * 32];
  const int qt = blockIdx.x, head = blockIdx.y;
  const int lane = threadIdx.x, half = lane >> 4, nc = lane & 15;
  const float scale = 0.08838834764831845f;
  v16h qf[4];
  const _Float16* qb = qp + (((size_t)head * (L / 16) + qt) * 4) * 512 + lane * 16;
  #pragma unroll
  for (int c = 0; c < 4; ++c) qf[c] = *(const v16h*)(qb + (size_t)c * 512);
  v8f zero = {0,0,0,0,0,0,0,0};
  v8f o[8];
  #pragma unroll
  for (int t = 0; t < 8; ++t) o[t] = zero;
  float mrow[8], lrow[8];
  #pragma unroll
  for (int r = 0; r < 8; ++r) { mrow[r] = -1e30f; lrow[r] = 0.f; }
  for (int kt = 0; kt < 8; kt += 2) {   // 128 keys = 8 tiles of 16
    v8f s0 = zero, s1 = zero;
    #pragma unroll
    for (int c = 0; c < 4; ++c) {
      v16h b0 = *(const v16h*)(akp + (((size_t)head * 8 + kt) * 4 + c) * 512 + lane * 16);
      v16h b1 = *(const v16h*)(akp + (((size_t)head * 8 + kt + 1) * 4 + c) * 512 + lane * 16);
      s0 = WMMA_F16(qf[c], b0, s0);
      s1 = WMMA_F16(qf[c], b1, s1);
    }
    const _Float16* vb = avp + (((size_t)head * 4 + (kt >> 1)) * 8) * 512 + lane * 16;
    v16h vf[8];
    #pragma unroll
    for (int t = 0; t < 8; ++t) vf[t] = *(const v16h*)(vb + (size_t)t * 512);
    float corr[8];
    #pragma unroll
    for (int r = 0; r < 8; ++r) {
      float a0 = s0[r] * scale, a1 = s1[r] * scale;
      float mx = fmaxf(a0, a1);
      #pragma unroll
      for (int off = 1; off < 16; off <<= 1) mx = fmaxf(mx, __shfl_xor(mx, off, 32));
      float nm = fmaxf(mrow[r], mx);
      corr[r] = __expf(mrow[r] - nm);
      float p0 = __expf(a0 - nm), p1 = __expf(a1 - nm);
      float ps = p0 + p1;
      #pragma unroll
      for (int off = 1; off < 16; off <<= 1) ps += __shfl_xor(ps, off, 32);
      lrow[r] = lrow[r] * corr[r] + ps;
      mrow[r] = nm;
      plds[(r + 8 * half) * 32 + nc]      = (_Float16)p0;
      plds[(r + 8 * half) * 32 + nc + 16] = (_Float16)p1;
    }
    __syncthreads();
    #pragma unroll
    for (int t = 0; t < 8; ++t)
      #pragma unroll
      for (int r = 0; r < 8; ++r) o[t][r] *= corr[r];
    v16h pf;
    #pragma unroll
    for (int p = 0; p < 8; ++p) {
      int k = kpair_base(p, half);
      pf[2 * p]     = plds[nc * 32 + k];
      pf[2 * p + 1] = plds[nc * 32 + k + 1];
    }
    #pragma unroll
    for (int t = 0; t < 8; ++t) o[t] = WMMA_F16(pf, vf[t], o[t]);
    __syncthreads();
  }
  float sc = fminf(ascale[0], 2.f);
  #pragma unroll
  for (int t = 0; t < 8; ++t) {
    int col = head * HD + t * 16 + nc;
    #pragma unroll
    for (int r = 0; r < 8; ++r) {
      int tok = qt * 16 + r + 8 * half;
      // face_mask (1,1,5,48,32) -> [:,0][:,:,::2,::2] -> (5,24,16) flat 1920
      int t5 = tok / 384, rr = (tok % 384) / 16, cc = tok % 16;
      float fm = fmask[t5 * 1536 + rr * 2 * 32 + cc * 2];
      attn[(size_t)tok * HID + col] += (o[t][r] / lrow[r]) * fm * sc;
    }
  }
}

// ---------------- final gating + residual ----------------
__global__ void final_kernel(const float* __restrict__ x, const float* __restrict__ outb,
                             const float* __restrict__ mod, const float* __restrict__ trm,
                             float* __restrict__ y) {
  int i = blockIdx.x * blockDim.x + threadIdx.x;
  if (i >= L * HID) return;
  int row = i / HID, c = i % HID;
  const float* mv = (row < FFN) ? trm : mod;
  y[i] = x[i] + outb[i] * mv[2 * HID + c];
}

// ---------------- launcher ----------------
extern "C" void kernel_launch(void* const* d_in, const int* in_sizes, int n_in,
                              void* d_out, int out_size, void* d_ws, size_t ws_size,
                              hipStream_t stream) {
  (void)in_sizes; (void)n_in; (void)out_size; (void)ws_size;
  const float* x      = (const float*)d_in[0];
  const float* vec    = (const float*)d_in[1];
  const float* trv    = (const float*)d_in[2];
  const float* fcos   = (const float*)d_in[3];
  const float* fsin   = (const float*)d_in[4];
  const float* aemb   = (const float*)d_in[5];
  const float* fmask  = (const float*)d_in[6];
  // d_in[7..9] = cu_seqlens_q / txt_len / first_frame_token_num (structure hardcoded)
  const float* mod_w  = (const float*)d_in[10];
  const float* mod_b  = (const float*)d_in[11];
  const float* w1     = (const float*)d_in[12];
  const float* b1     = (const float*)d_in[13];
  const float* qn_w   = (const float*)d_in[14];
  const float* kn_w   = (const float*)d_in[15];
  const float* w2     = (const float*)d_in[16];
  const float* b2     = (const float*)d_in[17];
  const float* aqkv_w = (const float*)d_in[18];
  const float* aqkv_b = (const float*)d_in[19];
  const float* akn_w  = (const float*)d_in[21];   // aud_qn_w unused (aq never consumed)
  const float* ascale = (const float*)d_in[22];
  float* y = (float*)d_out;

  char* p = (char*)d_ws;
  auto alloc = [&](size_t b) -> void* { void* r = p; p += (b + 255) & ~(size_t)255; return r; };
  float*    MODV = (float*)alloc((size_t)AQN * 4);
  float*    TRMV = (float*)alloc((size_t)AQN * 4);
  float*    XM   = (float*)alloc((size_t)L * HID * 4);
  _Float16* XMP  = (_Float16*)alloc((size_t)L * HID * 2);
  _Float16* W1P  = (_Float16*)alloc((size_t)HID * W1N * 2);
  float*    H    = (float*)alloc((size_t)L * W1N * 4);
  float*    RQ   = (float*)alloc((size_t)L * HEADS * 4);
  float*    RK   = (float*)alloc((size_t)L * HEADS * 4);
  _Float16* QP   = (_Float16*)alloc((size_t)HEADS * L * HD * 2);
  _Float16* KP   = (_Float16*)alloc((size_t)HEADS * L * HD * 2);
  _Float16* VP   = (_Float16*)alloc((size_t)HEADS * L * HD * 2);
  float*    ATTN = (float*)alloc((size_t)L * HID * 4);
  float*    AN   = (float*)alloc((size_t)L_AUD * HID * 4);
  _Float16* ANP  = (_Float16*)alloc((size_t)L_AUD * HID * 2);
  _Float16* AWP  = (_Float16*)alloc((size_t)HID * AQN * 2);
  float*    AQKV = (float*)alloc((size_t)L_AUD * AQN * 4);
  float*    ARK  = (float*)alloc((size_t)L_AUD * HEADS * 4);
  _Float16* AKP  = (_Float16*)alloc((size_t)HEADS * L_AUD * HD * 2);
  _Float16* AVP  = (_Float16*)alloc((size_t)HEADS * L_AUD * HD * 2);
  _Float16* A2P  = (_Float16*)alloc((size_t)L * K2 * 2);
  _Float16* W2P  = (_Float16*)alloc((size_t)K2 * HID * 2);
  float*    OUTB = (float*)alloc((size_t)L * HID * 4);

  dim3 b32(32), b256(256);

  // modulation vectors + modulated LN input
  mod_kernel<<<AQN / 256, b256, 0, stream>>>(vec, trv, mod_w, mod_b, MODV, TRMV);
  ln_mod_kernel<<<L, b256, 0, stream>>>(x, MODV, TRMV, XM);

  // GEMM1: h = xm @ w1 + b1
  pack_a_kernel<<<(L / 16) * (HID / 32), b32, 0, stream>>>(XM, XMP, HID);
  pack_b_kernel<<<(W1N / 16) * (HID / 32), b32, 0, stream>>>(w1, W1P, HID, W1N);
  wmma_gemm_kernel<<<dim3(W1N / 64, L / 32), b32, 0, stream>>>(XMP, W1P, b1, H, W1N, HID);

  // q/k RMS stats + fragment packing (rope fused)
  rms_stats_kernel<<<(L * HEADS + 255) / 256, b256, 0, stream>>>(H, (size_t)W1N, 0, L, RQ);
  rms_stats_kernel<<<(L * HEADS + 255) / 256, b256, 0, stream>>>(H, (size_t)W1N, HID, L, RK);
  pack_qk_kernel<<<HEADS * (L / 16) * 4, b32, 0, stream>>>(H, (size_t)W1N, 0, RQ, qn_w, fcos, fsin, L / 16, L_IMG, QP);
  pack_qk_kernel<<<HEADS * (L / 16) * 4, b32, 0, stream>>>(H, (size_t)W1N, HID, RK, kn_w, fcos, fsin, L / 16, L_IMG, KP);
  pack_v_kernel<<<HEADS * (L / 32) * 8, b32, 0, stream>>>(H, (size_t)W1N, 2 * HID, L / 32, VP);

  // self attention
  attn_kernel<<<dim3(L / 16, HEADS), b32, 0, stream>>>(QP, KP, VP, ATTN);

  // audio path: LN -> GEMM -> k-rms -> pack -> cross attention (adds into ATTN)
  ln_kernel<<<L_AUD, b256, 0, stream>>>(aemb, AN, HID);
  pack_a_kernel<<<(L_AUD / 16) * (HID / 32), b32, 0, stream>>>(AN, ANP, HID);
  pack_b_kernel<<<(AQN / 16) * (HID / 32), b32, 0, stream>>>(aqkv_w, AWP, HID, AQN);
  wmma_gemm_kernel<<<dim3(AQN / 64, L_AUD / 32), b32, 0, stream>>>(ANP, AWP, aqkv_b, AQKV, AQN, HID);
  rms_stats_kernel<<<(L_AUD * HEADS + 255) / 256, b256, 0, stream>>>(AQKV, (size_t)AQN, HID, L_AUD, ARK);
  pack_qk_kernel<<<HEADS * (L_AUD / 16) * 4, b32, 0, stream>>>(AQKV, (size_t)AQN, HID, ARK, akn_w, fcos, fsin, L_AUD / 16, 0, AKP);
  pack_v_kernel<<<HEADS * (L_AUD / 32) * 8, b32, 0, stream>>>(AQKV, (size_t)AQN, 2 * HID, L_AUD / 32, AVP);
  audio_attn_kernel<<<dim3(L_IMG / 16, HEADS), b32, 0, stream>>>(QP, AKP, AVP, fmask, ascale, ATTN);

  // GEMM2: out = [attn | gelu(mlp)] @ w2 + b2, then gate + residual
  pack_a2_kernel<<<(L / 16) * (K2 / 32), b32, 0, stream>>>(ATTN, H, A2P);
  pack_b_kernel<<<(HID / 16) * (K2 / 32), b32, 0, stream>>>(w2, W2P, K2, HID);
  wmma_gemm_kernel<<<dim3(HID / 64, L / 32), b32, 0, stream>>>(A2P, W2P, b2, OUTB, HID, K2);
  final_kernel<<<(L * HID + 255) / 256, b256, 0, stream>>>(x, OUTB, MODV, TRMV, y);
}
